// HMM_EM_28501402976724
// MI455X (gfx1250) — matrix operand order repaired and checked by reading, not compiled, not run on hardware
//
#include <hip/hip_runtime.h>
#include <hip/hip_bf16.h>

typedef __bf16 bf16_t;
typedef bf16_t v16bf __attribute__((ext_vector_type(16)));
typedef float  v8f   __attribute__((ext_vector_type(8)));

#define Z_DIM 256
#define X_DIM 64
#define SEQ_LEN 12
#define BATCH 8192

// ---------------- workspace layout (bytes) ----------------
// [0      .. 1023 ]  tmax[256]  f32   (col max of T_logits, softmax axis 0)
// [1024   .. 2047 ]  tsum[256]  f32
// [2048   .. 3071 ]  emax[256]  f32   (col max of emit_logits)
// [3072   .. 4095 ]  esum[256]  f32
// [4096   .. 5119 ]  pi[256]    f32
// [5120   .. 136191] T bf16, WMMA-B swizzled: 128 frags * 512 halves
// [136192 .. 168959] emit bf16 row-major 64x256
#define WS_T_OFF   5120
#define WS_E_OFF   (5120 + 131072)

// LDS layout (dynamic, bytes)
#define LDS_BETA_OFF 0                         // 128 rows * 256 f32 = 128 KB
#define LDS_ABUF_OFF (128 * 256 * 4)           // 64 frags * 512 bf16 = 64 KB
#define LDS_EMIT_OFF (LDS_ABUF_OFF + 64 * 512 * 2)  // 64*256 bf16 = 32 KB
#define LDS_TOTAL    (LDS_EMIT_OFF + X_DIM * Z_DIM * 2)

// ---------------- prep: column softmax stats + pi ----------------
__global__ void hmm_prep_cols(const float* __restrict__ Tl,
                              const float* __restrict__ pil,
                              const float* __restrict__ el,
                              float* __restrict__ ws) {
    int j = threadIdx.x;                       // 256 threads, 1 block
    float mx = -INFINITY;
    for (int k = 0; k < Z_DIM; ++k) mx = fmaxf(mx, Tl[k * Z_DIM + j]);
    float s = 0.f;
    for (int k = 0; k < Z_DIM; ++k) s += __expf(Tl[k * Z_DIM + j] - mx);
    ws[j] = mx; ws[256 + j] = s;
    float me = -INFINITY;
    for (int x = 0; x < X_DIM; ++x) me = fmaxf(me, el[x * Z_DIM + j]);
    float se = 0.f;
    for (int x = 0; x < X_DIM; ++x) se += __expf(el[x * Z_DIM + j] - me);
    ws[512 + j] = me; ws[768 + j] = se;
    float mp = -INFINITY;
    for (int z = 0; z < Z_DIM; ++z) mp = fmaxf(mp, pil[z]);
    float sp = 0.f;
    for (int z = 0; z < Z_DIM; ++z) sp += __expf(pil[z] - mp);
    ws[1024 + j] = __expf(pil[j] - mp) / sp;
}

// ---------------- prep: pack T into per-lane WMMA-B fragment order ------
// Fragment f = nt*8 + kt covers K in [kt*32, kt*32+32), N in [nt*16, nt*16+16).
// Contiguous-K B packing (not AMD's table order): lanes 0-15 hold K=0..15 (N=lane),
// lanes 16-31 hold K=16..31 (N=lane-16); 16 halves contiguous per lane.
__global__ void hmm_pack_T(const float* __restrict__ Tl,
                           const float* __restrict__ ws,
                           bf16_t* __restrict__ wsT) {
    int f = blockIdx.x;                        // 128 frags
    int lane = threadIdx.x;                    // 32
    int nt = f >> 3, kt = f & 7;
    int N  = nt * 16 + (lane & 15);
    int kb = kt * 32 + ((lane >> 4) << 4);     // 0 or 16 within the K-tile
    float mx  = ws[N];
    float inv = 1.0f / ws[256 + N];
    bf16_t* dst = wsT + (size_t)f * 512 + lane * 16;
    #pragma unroll
    for (int e = 0; e < 16; ++e) {
        int K = kb + e;
        dst[e] = (bf16_t)(__expf(Tl[K * Z_DIM + N] - mx) * inv);
    }
}

// ---------------- prep: pack emit (bf16 row-major 64x256) ----------------
__global__ void hmm_pack_emit(const float* __restrict__ el,
                              const float* __restrict__ ws,
                              bf16_t* __restrict__ wsE) {
    int x = blockIdx.x;                        // 64
    int z = threadIdx.x;                       // 256
    float v = __expf(el[x * Z_DIM + z] - ws[512 + z]) / ws[768 + z];
    wsE[x * Z_DIM + z] = (bf16_t)v;
}

// ---------------- main kernel ------------------------------------------
// 64 blocks x 256 threads (8 waves). The workgroup cooperatively owns a
// 128-row batch tile. "Weight-stationary" split: wave w pins the 16 B
// fragments of N-tiles {2w, 2w+1} in 128 VGPRs for the whole recursion
// (T read from L2 exactly once per workgroup). Per step:
//   phase 1: wave w builds bf16 A fragments (e[t+1] * beta) for M-tile w
//            from the LDS beta buffer and publishes them in an LDS A-buffer
//            (already in per-lane WMMA-A order -> 32B/lane ds_load later).
//   phase 2: every wave sweeps all 8 M-tiles: load A frag, 2 wmma into the
//            wave's 2 N-tile accumulators, write results back to beta LDS.
__global__ void
__launch_bounds__(256)
hmm_beta_kernel(const int* __restrict__ tokens,
                const bf16_t* __restrict__ wsT,
                const bf16_t* __restrict__ wsE,
                const float* __restrict__ pi,
                float* __restrict__ out) {
    extern __shared__ unsigned char smem[];
    float*  betaS = (float*)(smem + LDS_BETA_OFF);   // [128][256] f32
    bf16_t* abuf  = (bf16_t*)(smem + LDS_ABUF_OFF);  // [8 mt][8 kt][32 ln][16]
    bf16_t* emitS = (bf16_t*)(smem + LDS_EMIT_OFF);  // [64][256] bf16

    const int tid    = threadIdx.x;
    const int lane   = tid & 31;
    const int wave   = tid >> 5;
    const int l15    = lane & 15;
    const int hiHalf = lane >> 4;                    // 0 or 1
    const int kh     = hiHalf << 3;                  // A-layout K sub-offset
    const int b0     = blockIdx.x * 128;             // block batch base
    const int myRow  = wave * 16 + l15;              // row this lane gathers

    // ---- pin this wave's 16 B fragments (2 N-tiles x 8 K-tiles) in VGPRs
    v16bf Bfr[16];
    #pragma unroll
    for (int j = 0; j < 2; ++j) {
        const int ntG = wave * 2 + j;
        #pragma unroll
        for (int kt = 0; kt < 8; ++kt)
            Bfr[j * 8 + kt] = *(const v16bf*)(wsT +
                ((size_t)(ntG * 8 + kt) * 512 + lane * 16));
    }

    // ---- stage emit table; init beta = 1 (each wave its own 16 rows)
    for (int i = tid; i < X_DIM * Z_DIM; i += 256) emitS[i] = wsE[i];
    for (int i = lane; i < 16 * 256; i += 32) betaS[wave * 16 * 256 + i] = 1.0f;
    __syncthreads();

    for (int t = SEQ_LEN - 2; t >= 0; --t) {
        // ---- phase 1: owner wave builds A = e[t+1] * beta for M-tile `wave`
        // A in contiguous K order (not AMD table order): lane row m = lane&15; lanes<16 cover
        // K {0..15}, lanes>=16 cover K {16..31} per K-tile (matches B pack).
        {
            int tok = tokens[(t + 1) * BATCH + b0 + myRow];
            const bf16_t* erow = emitS + tok * Z_DIM;
            const float*  brow = betaS + myRow * 256;
            #pragma unroll
            for (int kt = 0; kt < 8; ++kt) {
                v16bf af;
                #pragma unroll
                for (int e = 0; e < 16; ++e) {
                    int K = kt * 32 + (hiHalf << 4) + e;
                    af[e] = (bf16_t)(brow[K] * (float)erow[K]);
                }
                *(v16bf*)(abuf + ((wave * 8 + kt) * 32 + lane) * 16) = af;
            }
        }
        __syncthreads();   // A published; beta consumed -> safe to overwrite

        // ---- phase 2: sweep all M-tiles with the pinned B fragments
        #pragma unroll
        for (int mt = 0; mt < 8; ++mt) {
            v8f c0 = {}, c1 = {};
            #pragma unroll
            for (int kt = 0; kt < 8; ++kt) {
                v16bf af = *(const v16bf*)(abuf +
                    ((mt * 8 + kt) * 32 + lane) * 16);
                c0 = __builtin_amdgcn_wmma_f32_16x16x32_bf16(
                         false, af, false, Bfr[kt],     (short)0, c0, false, false);
                c1 = __builtin_amdgcn_wmma_f32_16x16x32_bf16(
                         false, af, false, Bfr[8 + kt], (short)0, c1, false, false);
            }
            // write back beta_new (C layout: lane->N, vgpr->M)
            const int n0 = (wave * 2) * 16 + l15;
            #pragma unroll
            for (int r = 0; r < 8; ++r) {
                int m = mt * 16 + r + (hiHalf << 3);
                betaS[m * 256 + n0]      = c0[r];
                betaS[m * 256 + n0 + 16] = c1[r];
            }
        }
        __syncthreads();   // beta_new complete before next step's A-build
    }

    // ---- epilogue: out[b] = -log( sum_z e0 * pi * beta0 )
    // alpha0*beta0 ~ 1e-24: inside f32 range, so plain sum + one log is
    // exactly logsumexp(log a + log b). Each lane does half a row.
    {
        int tok0 = tokens[b0 + myRow];
        const bf16_t* erow = emitS + tok0 * Z_DIM;
        const float*  brow = betaS + myRow * 256;
        float s = 0.f;
        #pragma unroll 4
        for (int zz = 0; zz < 128; ++zz) {
            int z = hiHalf * 128 + zz;
            s += brow[z] * (float)erow[z] * pi[z];
        }
        s += __shfl_xor(s, 16, 32);
        if (lane < 16) out[b0 + myRow] = -__logf(s);
    }
}

extern "C" void kernel_launch(void* const* d_in, const int* in_sizes, int n_in,
                              void* d_out, int out_size, void* d_ws, size_t ws_size,
                              hipStream_t stream) {
    const int*   tokens = (const int*)d_in[0];   // (12, 8192) int32
    const float* Tl     = (const float*)d_in[1]; // (256, 256) f32
    const float* pil    = (const float*)d_in[2]; // (256,) f32
    const float* el     = (const float*)d_in[3]; // (64, 256) f32
    float* out = (float*)d_out;                  // (8192,) f32

    float*  wsf = (float*)d_ws;
    bf16_t* wsT = (bf16_t*)((char*)d_ws + WS_T_OFF);
    bf16_t* wsE = (bf16_t*)((char*)d_ws + WS_E_OFF);

    hmm_prep_cols<<<1, 256, 0, stream>>>(Tl, pil, el, wsf);
    hmm_pack_T   <<<128, 32, 0, stream>>>(Tl, wsf, wsT);
    hmm_pack_emit<<<64, 256, 0, stream>>>(el, wsf, wsE);

    hmm_beta_kernel<<<BATCH / 128, 256, LDS_TOTAL, stream>>>(
        tokens, wsT, wsE, wsf + 1024, out);
}
